// Linearized_Softmax_Attention_14448269984165
// MI455X (gfx1250) — matrix operand order repaired and checked by reading, not compiled, not run on hardware
//
#include <hip/hip_runtime.h>

#define BATCH 8192
#define INSZ  8192
#define LTOK  64
#define DEMB  128
#define LDT   72      // padded leading dim for K^T / V^T (72*2B=144B -> conflict-free stride)

#if defined(__has_builtin)
#if __has_builtin(__builtin_amdgcn_global_load_async_to_lds_b128) && \
    __has_builtin(__builtin_amdgcn_s_wait_asynccnt)
#define USE_ASYNC_LDS 1
#endif
#endif

typedef __attribute__((ext_vector_type(16))) __bf16 v16bf;
typedef __attribute__((ext_vector_type(8)))  __bf16 v8bf;
typedef __attribute__((ext_vector_type(8)))  float  v8f;
typedef __attribute__((ext_vector_type(4)))  int    v4i;

union Frag16 {
    v16bf v;
    unsigned short s[16];
    v8bf h[2];
};

__device__ __forceinline__ unsigned short f2bf(float f) {
    unsigned int u = __builtin_bit_cast(unsigned int, f);
    unsigned int r = u + 0x7FFFu + ((u >> 16) & 1u);   // round-to-nearest-even
    return (unsigned short)(r >> 16);
}
__device__ __forceinline__ float bf2f(unsigned short h) {
    unsigned int u = ((unsigned int)h) << 16;
    return __builtin_bit_cast(float, u);
}

// ---- fragment loaders (gfx1250 wave32 WMMA VGPR layouts, CDNA5 ISA) ----

// A fragment, 16x32 bf16, source row-major [.. x ldm], tile origin (r0,k0).
// lane 0-15: rows, K = {0..7, 16..23}; lane 16-31: same rows, K = {8..15, 24..31}.
__device__ __forceinline__ void load_a(Frag16& a, const unsigned short* mat, int ldm,
                                       int r0, int k0, int lane) {
    int g = (lane >> 4) & 1, m = lane & 15;
    const unsigned short* p = mat + (r0 + m) * ldm + k0 + g * 8;
    a.h[0] = *(const v8bf*)(p);        // K = g*8 .. g*8+7
    a.h[1] = *(const v8bf*)(p + 16);   // K = 16+g*8 .. 16+g*8+7
}

// B fragment, 32x16 bf16, from TRANSPOSED source: B(k,n) = mat[(n0+n)*ldm + k0+k].
// lane layout (lane=n, K pairs per VGPR) -> two contiguous 16B LDS loads.
__device__ __forceinline__ void load_b_t(Frag16& b, const unsigned short* mat, int ldm,
                                         int k0, int n0, int lane) {
    int g = (lane >> 4) & 1, n = lane & 15;
    const unsigned short* p = mat + (n0 + n) * ldm + k0 + g * 8;
    b.h[0] = *(const v8bf*)(p);
    b.h[1] = *(const v8bf*)(p + 16);
}

__device__ __forceinline__ v8f wmma_bf16(const Frag16& a, const Frag16& b, v8f c) {
    return __builtin_amdgcn_wmma_f32_16x16x32_bf16(false, a.v, false, b.v,
                                                   (short)0, c, false, false);
}

// ---- pre-pass: fp32 weights -> bf16 in workspace ----
__global__ __launch_bounds__(256) void convert_w_kernel(const float* __restrict__ wq,
                                                        const float* __restrict__ wk,
                                                        const float* __restrict__ wv,
                                                        unsigned short* __restrict__ dst) {
    int i = blockIdx.x * 256 + threadIdx.x;               // 0 .. 3*16384-1
    const float* src = (i < 16384) ? wq : (i < 32768) ? wk : wv;
    dst[i] = f2bf(src[i & 16383]);
}

// ---- main: one workgroup per batch row ----
__global__ __launch_bounds__(256) void lin_attn_kernel(
    const float* __restrict__ x,
    const unsigned short* __restrict__ wbf,   // [3][128*128] bf16: Wq,Wk,Wv
    const float* __restrict__ bq, const float* __restrict__ bk, const float* __restrict__ bv,
    const float* __restrict__ lnw, const float* __restrict__ lnb,
    float* __restrict__ out)
{
    __shared__ alignas(32) unsigned short sXn[LTOK * DEMB];   // 16 KB layernormed x (bf16)
    __shared__ alignas(32) unsigned short sW [DEMB * DEMB];   // 32 KB weight buf, reused as KV
    __shared__ alignas(32) unsigned short sQ [LTOK * DEMB];   // 16 KB  Q row-major
    __shared__ alignas(32) unsigned short sKt[DEMB * LDT];    // 18 KB  K^T [d][i], padded
    __shared__ alignas(32) unsigned short sVt[DEMB * LDT];    // 18 KB  V^T [d][i], padded
    __shared__ float sKsum[DEMB];
    __shared__ float sNorm[LTOK];
    __shared__ float sRed[16];
    __shared__ float sStat[2];

    const int tid  = threadIdx.x;
    const int lane = tid & 31;
    const int wave = tid >> 5;
    const long long base = (long long)blockIdx.x * INSZ;

    // ---------------- Phase 1: LayerNorm over 8192 features ----------------
    float xr[32];
    {
        const float4* xv = (const float4*)(x + base);
        float s = 0.f, sq = 0.f;
#pragma unroll
        for (int j = 0; j < 8; ++j) {
            float4 t = xv[tid + 256 * j];
            xr[4*j+0] = t.x; xr[4*j+1] = t.y; xr[4*j+2] = t.z; xr[4*j+3] = t.w;
            s  += t.x + t.y + t.z + t.w;
            sq += t.x*t.x + t.y*t.y + t.z*t.z + t.w*t.w;
        }
#pragma unroll
        for (int off = 16; off; off >>= 1) {
            s  += __shfl_xor(s,  off, 32);
            sq += __shfl_xor(sq, off, 32);
        }
        if (lane == 0) { sRed[wave] = s; sRed[8 + wave] = sq; }
        __syncthreads();
        if (tid == 0) {
            float ts = 0.f, tq = 0.f;
#pragma unroll
            for (int i = 0; i < 8; ++i) { ts += sRed[i]; tq += sRed[8 + i]; }
            float mu  = ts * (1.0f / INSZ);
            float var = tq * (1.0f / INSZ) - mu * mu;
            sStat[0] = mu;
            sStat[1] = rsqrtf(var + 1e-5f);
        }
        __syncthreads();
        float mu = sStat[0], rs = sStat[1];
        const float4* wv4 = (const float4*)lnw;
        const float4* bv4 = (const float4*)lnb;
#pragma unroll
        for (int j = 0; j < 8; ++j) {
            int e4 = tid + 256 * j;
            float4 w = wv4[e4], b = bv4[e4];
            int e0 = e4 * 4;
            sXn[e0 + 0] = f2bf((xr[4*j+0] - mu) * rs * w.x + b.x);
            sXn[e0 + 1] = f2bf((xr[4*j+1] - mu) * rs * w.y + b.y);
            sXn[e0 + 2] = f2bf((xr[4*j+2] - mu) * rs * w.z + b.z);
            sXn[e0 + 3] = f2bf((xr[4*j+3] - mu) * rs * w.w + b.w);
        }
    }
    __syncthreads();

    // ---------------- Phase 2: Q = elu(X Wq^T + bq)+1, K likewise (->K^T), V (->V^T) ----
    // Output [64x128] = 4x8 tiles of 16x16; each wave owns 4 tiles; K-dim 128 = 4 chunks.
    for (int gm = 0; gm < 3; ++gm) {
        __syncthreads();   // previous readers of sW done
#if USE_ASYNC_LDS
        {
            // async DMA of this 32KB weight tile into LDS (ASYNCcnt-tracked)
            auto g1 = (__attribute__((address_space(1))) v4i*)(wbf + gm * DEMB * DEMB + tid * 8);
            auto l3 = (__attribute__((address_space(3))) v4i*)(sW + tid * 8);
            __builtin_amdgcn_global_load_async_to_lds_b128(g1, l3, 0,     0);
            __builtin_amdgcn_global_load_async_to_lds_b128(g1, l3, 4096,  0);
            __builtin_amdgcn_global_load_async_to_lds_b128(g1, l3, 8192,  0);
            __builtin_amdgcn_global_load_async_to_lds_b128(g1, l3, 12288, 0);
            __builtin_amdgcn_global_load_async_to_lds_b128(g1, l3, 16384, 0);
            __builtin_amdgcn_global_load_async_to_lds_b128(g1, l3, 20480, 0);
            __builtin_amdgcn_global_load_async_to_lds_b128(g1, l3, 24576, 0);
            __builtin_amdgcn_global_load_async_to_lds_b128(g1, l3, 28672, 0);
            __builtin_amdgcn_s_wait_asynccnt(0);
        }
#else
        {
            const uint4* src = (const uint4*)(wbf + gm * DEMB * DEMB);
            uint4* dst = (uint4*)sW;
#pragma unroll
            for (int j = 0; j < 8; ++j) dst[tid + 256 * j] = src[tid + 256 * j];
        }
#endif
        __syncthreads();
        const float* bias = (gm == 0) ? bq : (gm == 1) ? bk : bv;
#pragma unroll
        for (int t = 0; t < 4; ++t) {
            int tile = wave * 4 + t;
            int tr = tile >> 3, tc = tile & 7;
            v8f acc = {0.f, 0.f, 0.f, 0.f, 0.f, 0.f, 0.f, 0.f};
            Frag16 a, bb;
#pragma unroll
            for (int kb = 0; kb < 4; ++kb) {
                load_a(a, sXn, DEMB, tr * 16, kb * 32, lane);           // X tile
                load_b_t(bb, sW, DEMB, kb * 32, tc * 16, lane);         // B(k,n)=W[n][k]
                acc = wmma_bf16(a, bb, acc);
            }
            int n = lane & 15, g = lane >> 4;
            float bsv = bias[tc * 16 + n];
            if (gm == 0) {
                // Q row-major, elu(v)+1
#pragma unroll
                for (int j = 0; j < 8; ++j) {
                    float v = acc[j] + bsv;
                    v = (v > 0.f) ? (v + 1.f) : __expf(v);
                    sQ[(tr * 16 + g * 8 + j) * DEMB + tc * 16 + n] = f2bf(v);
                }
            } else {
                // K^T / V^T: 8 consecutive rows per lane -> one packed 16B LDS store
                unsigned short* dt = (gm == 1) ? sKt : sVt;
                union { unsigned short us[8]; uint4 u4; } pk;
#pragma unroll
                for (int j = 0; j < 8; ++j) {
                    float v = acc[j] + bsv;
                    if (gm == 1) v = (v > 0.f) ? (v + 1.f) : __expf(v);  // elu(v)+1 for K
                    pk.us[j] = f2bf(v);
                }
                *(uint4*)(dt + (tc * 16 + n) * LDT + tr * 16 + g * 8) = pk.u4;
            }
        }
    }
    __syncthreads();

    // bring x back toward the caches for the residual read in phase 4
    __builtin_prefetch(x + base + tid * 32, 0, 3);

    // ---------------- Phase 3: Ksum, KV = V^T K (into sW), norm ----------------
    if (tid < DEMB) {
        const unsigned int* p = (const unsigned int*)(sKt + tid * LDT);
        float s = 0.f;
#pragma unroll 8
        for (int i = 0; i < LTOK / 2; ++i) {
            unsigned int u = p[i];
            s += bf2f((unsigned short)u) + bf2f((unsigned short)(u >> 16));
        }
        sKsum[tid] = s;
    }
    // KV [128x128] = 8x8 tiles; each wave owns 8; K-dim 64 tokens = 2 chunks.
#pragma unroll
    for (int t = 0; t < 8; ++t) {
        int tile = wave * 8 + t;
        int tr = tile >> 3, tc = tile & 7;
        v8f acc = {0.f, 0.f, 0.f, 0.f, 0.f, 0.f, 0.f, 0.f};
        Frag16 a, bb;
#pragma unroll
        for (int kb = 0; kb < 2; ++kb) {
            load_a(a, sVt, LDT, tr * 16, kb * 32, lane);     // A(m,k)=Vt[vdim][i]=V[i][vdim]
            load_b_t(bb, sKt, LDT, kb * 32, tc * 16, lane);  // B(k,n)=Kt[kdim][i]^T=K[i][kdim]
            acc = wmma_bf16(a, bb, acc);
        }
        int n = lane & 15, g = lane >> 4;
#pragma unroll
        for (int j = 0; j < 8; ++j)
            sW[(tr * 16 + g * 8 + j) * DEMB + tc * 16 + n] = f2bf(acc[j]);
    }
    __syncthreads();
    if (tid < LTOK) {
        const unsigned int* p = (const unsigned int*)(sQ + tid * DEMB);
        float s = 0.f;
#pragma unroll 8
        for (int d = 0; d < DEMB / 2; ++d) {
            unsigned int u = p[d];
            s += bf2f((unsigned short)u) * sKsum[2 * d]
               + bf2f((unsigned short)(u >> 16)) * sKsum[2 * d + 1];
        }
        sNorm[tid] = 1.f / (s + 1e-7f);
    }
    __syncthreads();

    // ---------------- Phase 4: out = norm[l] * (Q KV^T)[l,m] + x ----------------
#pragma unroll
    for (int t = 0; t < 4; ++t) {
        int tile = wave * 4 + t;
        int tr = tile >> 3, tc = tile & 7;
        v8f acc = {0.f, 0.f, 0.f, 0.f, 0.f, 0.f, 0.f, 0.f};
        Frag16 a, bb;
#pragma unroll
        for (int kb = 0; kb < 4; ++kb) {
            load_a(a, sQ, DEMB, tr * 16, kb * 32, lane);
            load_b_t(bb, sW, DEMB, kb * 32, tc * 16, lane);  // B(k,n)=KV[m][d]
            acc = wmma_bf16(a, bb, acc);
        }
        int n = lane & 15, g = lane >> 4;
#pragma unroll
        for (int j = 0; j < 8; ++j) {
            int row = tr * 16 + g * 8 + j;
            int col = tc * 16 + n;
            long long idx = base + row * DEMB + col;
            out[idx] = acc[j] * sNorm[row] + x[idx];
        }
    }
}

extern "C" void kernel_launch(void* const* d_in, const int* in_sizes, int n_in,
                              void* d_out, int out_size, void* d_ws, size_t ws_size,
                              hipStream_t stream) {
    const float* x   = (const float*)d_in[0];
    const float* Wq  = (const float*)d_in[1];
    const float* bq  = (const float*)d_in[2];
    const float* Wk  = (const float*)d_in[3];
    const float* bk  = (const float*)d_in[4];
    const float* Wv  = (const float*)d_in[5];
    const float* bv  = (const float*)d_in[6];
    const float* lnw = (const float*)d_in[7];
    const float* lnb = (const float*)d_in[8];
    float* out = (float*)d_out;
    unsigned short* wbf = (unsigned short*)d_ws;   // 3 * 128*128 bf16 = 96 KB

    convert_w_kernel<<<192, 256, 0, stream>>>(Wq, Wk, Wv, wbf);
    lin_attn_kernel<<<BATCH, 256, 0, stream>>>(x, wbf, bq, bk, bv, lnw, lnb, out);
}
